// AttentionLayer_77730318123665
// MI455X (gfx1250) — matrix operand order repaired and checked by reading, not compiled, not run on hardware
//
#include <hip/hip_runtime.h>
#include <hip/hip_bf16.h>
#include <math.h>

// ---------------------------------------------------------------------------
// Attention layer on MI455X (gfx1250, wave32, WMMA).
//   B=4, S=2048, D=512.  ~47 GFLOP, ~140MB working set -> fits global L2
//   (192MB). Compute-bound => v_wmma_f32_16x16x32_bf16 everywhere.
// All WMMA operands load per lane as two b128 pieces in AMD table k order: A row-major,
// B always through the transposed loader (weights pre-transposed, V written
// transposed by the projection kernel).
// Register blocking: projections 2x2 (32x32/wave); scores and attn@V GEMMs
// 4x2 (64x32/wave) -> 8 WMMAs per 12 b128 loads per k-step.
// ---------------------------------------------------------------------------

#define BB 4
#define SS 2048
#define DD 512

typedef __attribute__((ext_vector_type(16))) __bf16 v16bf;
typedef __attribute__((ext_vector_type(8)))  float  v8f;

#define WMMA_BF16(a, b, c) \
  __builtin_amdgcn_wmma_f32_16x16x32_bf16(false, (a), false, (b), (short)0, (c), false, false)

// ---- fragment load/store helpers (ISA 7.12.2 layouts, wave32) -------------

__device__ __forceinline__ v16bf load2x16B(const __bf16* p0, const __bf16* p1) {
  union { v16bf v; uint4 q[2]; } u;
  u.q[0] = *reinterpret_cast<const uint4*>(p0);
  u.q[1] = *reinterpret_cast<const uint4*>(p1);
  return u.v;
}

// A: 16x32 bf16 row-major, p -> (m0,k0), stride ld.
__device__ __forceinline__ v16bf load_frag_a(const __bf16* p, int ld) {
  const int lane = threadIdx.x & 31;
  const __bf16* row = p + (size_t)(lane & 15) * ld + ((lane >> 4) << 3);
  return load2x16B(row, row + 16);
}

// B = src^T: logical B[k][n] = src[n][k]; src row-major, p -> (n0,k0).
__device__ __forceinline__ v16bf load_frag_bT(const __bf16* p, int ld) {
  const int lane = threadIdx.x & 31;
  const __bf16* row = p + (size_t)(lane & 15) * ld + ((lane >> 4) << 3);
  return load2x16B(row, row + 16);
}

// C/D 16x16 f32: VGPR j -> row M = j + 8*(lane>=16), col N = lane&15.
__device__ __forceinline__ void store_c_f32(float* p, int ld, const v8f& c) {
  const int lane  = threadIdx.x & 31;
  const int n     = lane & 15;
  const int mbase = (lane >> 4) << 3;
#pragma unroll
  for (int j = 0; j < 8; ++j) p[(size_t)(mbase + j) * ld + n] = c[j];
}

// row-major bf16 store with bias (for Q,K).
__device__ __forceinline__ void store_c_bf16_bias(__bf16* p, int ld, const v8f& c,
                                                  const float* bias16) {
  const int lane  = threadIdx.x & 31;
  const int n     = lane & 15;
  const int mbase = (lane >> 4) << 3;
  const float bv  = bias16[n];
#pragma unroll
  for (int j = 0; j < 8; ++j)
    p[(size_t)(mbase + j) * ld + n] = (__bf16)(c[j] + bv);
}

// TRANSPOSED bf16 store with bias (for V^T): one 16B store per lane.
__device__ __forceinline__ void store_c_bf16_bias_T(__bf16* p, int ldT, const v8f& c,
                                                    const float* bias16) {
  const int lane  = threadIdx.x & 31;
  const int n     = lane & 15;
  const int mbase = (lane >> 4) << 3;
  const float bv  = bias16[n];
  union { __bf16 h[8]; uint4 q; } u;
#pragma unroll
  for (int j = 0; j < 8; ++j) u.h[j] = (__bf16)(c[j] + bv);
  *reinterpret_cast<uint4*>(p + (size_t)n * ldT + mbase) = u.q;
}

// ---- k0a: fp32 -> bf16 convert (x) ----------------------------------------
__global__ __launch_bounds__(256) void k_cvt_bf16(const float* __restrict__ in,
                                                  __bf16* __restrict__ out, int n) {
  int i = blockIdx.x * 256 + threadIdx.x;
  if (i < n) out[i] = (__bf16)in[i];
}

// ---- k0b: fp32 -> bf16 convert + transpose (weights, DDxDD) ---------------
__global__ __launch_bounds__(256) void k_cvt_bf16_T(const float* __restrict__ in,
                                                    __bf16* __restrict__ outT) {
  int i = blockIdx.x * 256 + threadIdx.x;        // i = d*DD + e
  int d = i >> 9, e = i & (DD - 1);
  outT[(size_t)e * DD + d] = (__bf16)in[i];
}

// ---- k1: projections, 2x2 blocked. M=B*S=8192, N=K=DD ---------------------
template <bool TSTORE>
__global__ __launch_bounds__(256) void k_proj(const __bf16* __restrict__ X,
                                              const __bf16* __restrict__ Wt,
                                              const float* __restrict__ bias,
                                              __bf16* __restrict__ C) {
  const int w  = blockIdx.x * 8 + (threadIdx.x >> 5);
  const int tn = (w & 15) << 5;                  // N/32 = 16 col blocks
  const int tm = (w >> 4) << 5;                  // row block
  const __bf16* a0 = X  + (size_t)tm * DD;
  const __bf16* a1 = a0 + (size_t)16 * DD;
  const __bf16* b0 = Wt + (size_t)tn * DD;
  const __bf16* b1 = b0 + (size_t)16 * DD;
  v8f c00 = {}, c01 = {}, c10 = {}, c11 = {};
#pragma unroll 2
  for (int k = 0; k < DD; k += 32) {
    __builtin_prefetch(a0 + k + 128, 0, 3);
    __builtin_prefetch(a1 + k + 128, 0, 3);
    v16bf A0 = load_frag_a(a0 + k, DD);
    v16bf A1 = load_frag_a(a1 + k, DD);
    v16bf B0 = load_frag_bT(b0 + k, DD);
    v16bf B1 = load_frag_bT(b1 + k, DD);
    c00 = WMMA_BF16(A0, B0, c00);
    c01 = WMMA_BF16(A0, B1, c01);
    c10 = WMMA_BF16(A1, B0, c10);
    c11 = WMMA_BF16(A1, B1, c11);
  }
  if (TSTORE) {
    const int b  = tm / SS;                      // 32-row block never crosses batch
    const int s0 = tm & (SS - 1);
    __bf16* ct = C + (size_t)b * DD * SS;
    store_c_bf16_bias_T(ct + (size_t)tn        * SS + s0,      SS, c00, bias + tn);
    store_c_bf16_bias_T(ct + (size_t)(tn + 16) * SS + s0,      SS, c01, bias + tn + 16);
    store_c_bf16_bias_T(ct + (size_t)tn        * SS + s0 + 16, SS, c10, bias + tn);
    store_c_bf16_bias_T(ct + (size_t)(tn + 16) * SS + s0 + 16, SS, c11, bias + tn + 16);
  } else {
    __bf16* cp = C + (size_t)tm * DD + tn;
    store_c_bf16_bias(cp,                        DD, c00, bias + tn);
    store_c_bf16_bias(cp + 16,                   DD, c01, bias + tn + 16);
    store_c_bf16_bias(cp + (size_t)16 * DD,      DD, c10, bias + tn);
    store_c_bf16_bias(cp + (size_t)16 * DD + 16, DD, c11, bias + tn + 16);
  }
}

// ---- k2: scores = Q @ K^T per batch, 4x2 blocked (64x32 per wave) ---------
__global__ __launch_bounds__(256) void k_scores(const __bf16* __restrict__ Q,
                                                const __bf16* __restrict__ Km,
                                                float* __restrict__ Sc) {
  const int b  = blockIdx.y;
  const int w  = blockIdx.x * 8 + (threadIdx.x >> 5);
  const int tn = (w & 63) << 5;                  // S/32 = 64 key blocks
  const int tm = (w >> 6) << 6;                  // S/64 = 32 query blocks
  const __bf16* qb = Q  + (size_t)b * SS * DD + (size_t)tm * DD;
  const __bf16* kb = Km + (size_t)b * SS * DD + (size_t)tn * DD;
  v8f c00 = {}, c01 = {}, c10 = {}, c11 = {};
  v8f c20 = {}, c21 = {}, c30 = {}, c31 = {};
#pragma unroll 2
  for (int k = 0; k < DD; k += 32) {
    __builtin_prefetch(qb + k + 128, 0, 3);
    __builtin_prefetch(kb + k + 128, 0, 3);
    v16bf A0 = load_frag_a(qb + k,                     DD);
    v16bf A1 = load_frag_a(qb + (size_t)16 * DD + k,   DD);
    v16bf A2 = load_frag_a(qb + (size_t)32 * DD + k,   DD);
    v16bf A3 = load_frag_a(qb + (size_t)48 * DD + k,   DD);
    v16bf B0 = load_frag_bT(kb + k,                    DD);
    v16bf B1 = load_frag_bT(kb + (size_t)16 * DD + k,  DD);
    c00 = WMMA_BF16(A0, B0, c00);
    c01 = WMMA_BF16(A0, B1, c01);
    c10 = WMMA_BF16(A1, B0, c10);
    c11 = WMMA_BF16(A1, B1, c11);
    c20 = WMMA_BF16(A2, B0, c20);
    c21 = WMMA_BF16(A2, B1, c21);
    c30 = WMMA_BF16(A3, B0, c30);
    c31 = WMMA_BF16(A3, B1, c31);
  }
  float* sp = Sc + (size_t)b * SS * SS + (size_t)tm * SS + tn;
  store_c_f32(sp,                          SS, c00);
  store_c_f32(sp + 16,                     SS, c01);
  store_c_f32(sp + (size_t)16 * SS,        SS, c10);
  store_c_f32(sp + (size_t)16 * SS + 16,   SS, c11);
  store_c_f32(sp + (size_t)32 * SS,        SS, c20);
  store_c_f32(sp + (size_t)32 * SS + 16,   SS, c21);
  store_c_f32(sp + (size_t)48 * SS,        SS, c30);
  store_c_f32(sp + (size_t)48 * SS + 16,   SS, c31);
}

// ---- k3: row softmax (row length SS), fp32 in, bf16 out -------------------
__global__ __launch_bounds__(256) void k_softmax(const float* __restrict__ Sc,
                                                 __bf16* __restrict__ P) {
  __shared__ float red[256];
  const size_t row = blockIdx.x;
  const float* src = Sc + row * SS;
  __bf16*      dst = P  + row * SS;
  const int t = threadIdx.x;
  float v[SS / 256];
  float m = -3.0e38f;
#pragma unroll
  for (int i = 0; i < SS / 256; ++i) {
    v[i] = src[t + 256 * i];
    m = fmaxf(m, v[i]);
  }
  red[t] = m;
  __syncthreads();
#pragma unroll
  for (int off = 128; off > 0; off >>= 1) {
    if (t < off) red[t] = fmaxf(red[t], red[t + off]);
    __syncthreads();
  }
  m = red[0];
  __syncthreads();
  float s = 0.f;
#pragma unroll
  for (int i = 0; i < SS / 256; ++i) {
    v[i] = __expf(v[i] - m);
    s += v[i];
  }
  red[t] = s;
  __syncthreads();
#pragma unroll
  for (int off = 128; off > 0; off >>= 1) {
    if (t < off) red[t] += red[t + off];
    __syncthreads();
  }
  const float inv = 1.0f / red[0];
#pragma unroll
  for (int i = 0; i < SS / 256; ++i)
    dst[t + 256 * i] = (__bf16)(v[i] * inv);
}

// ---- k4: out = attn @ V per batch, 4x2 blocked; V stored transposed -------
__global__ __launch_bounds__(256) void k_out(const __bf16* __restrict__ P,
                                             const __bf16* __restrict__ Vt,
                                             float* __restrict__ O) {
  const int b  = blockIdx.y;
  const int w  = blockIdx.x * 8 + (threadIdx.x >> 5);
  const int tn = (w & 15) << 5;                  // D/32 = 16 col blocks
  const int tm = (w >> 4) << 6;                  // S/64 = 32 query blocks
  const __bf16* pb = P  + (size_t)b * SS * SS + (size_t)tm * SS;
  const __bf16* vb = Vt + (size_t)b * DD * SS + (size_t)tn * SS;
  v8f c00 = {}, c01 = {}, c10 = {}, c11 = {};
  v8f c20 = {}, c21 = {}, c30 = {}, c31 = {};
#pragma unroll 2
  for (int k = 0; k < SS; k += 32) {
    __builtin_prefetch(pb + k + 128, 0, 3);
    __builtin_prefetch(vb + k + 128, 0, 3);
    v16bf A0 = load_frag_a(pb + k,                    SS);
    v16bf A1 = load_frag_a(pb + (size_t)16 * SS + k,  SS);
    v16bf A2 = load_frag_a(pb + (size_t)32 * SS + k,  SS);
    v16bf A3 = load_frag_a(pb + (size_t)48 * SS + k,  SS);
    v16bf B0 = load_frag_bT(vb + k,                   SS);
    v16bf B1 = load_frag_bT(vb + (size_t)16 * SS + k, SS);
    c00 = WMMA_BF16(A0, B0, c00);
    c01 = WMMA_BF16(A0, B1, c01);
    c10 = WMMA_BF16(A1, B0, c10);
    c11 = WMMA_BF16(A1, B1, c11);
    c20 = WMMA_BF16(A2, B0, c20);
    c21 = WMMA_BF16(A2, B1, c21);
    c30 = WMMA_BF16(A3, B0, c30);
    c31 = WMMA_BF16(A3, B1, c31);
  }
  float* op = O + (size_t)b * SS * DD + (size_t)tm * DD + tn;
  store_c_f32(op,                          DD, c00);
  store_c_f32(op + 16,                     DD, c01);
  store_c_f32(op + (size_t)16 * DD,        DD, c10);
  store_c_f32(op + (size_t)16 * DD + 16,   DD, c11);
  store_c_f32(op + (size_t)32 * DD,        DD, c20);
  store_c_f32(op + (size_t)32 * DD + 16,   DD, c21);
  store_c_f32(op + (size_t)48 * DD,        DD, c30);
  store_c_f32(op + (size_t)48 * DD + 16,   DD, c31);
}

// ---------------------------------------------------------------------------
extern "C" void kernel_launch(void* const* d_in, const int* in_sizes, int n_in,
                              void* d_out, int out_size, void* d_ws, size_t ws_size,
                              hipStream_t stream) {
  const float* x  = (const float*)d_in[0];
  const float* Wq = (const float*)d_in[1];
  const float* Wk = (const float*)d_in[2];
  const float* Wv = (const float*)d_in[3];
  const float* bq = (const float*)d_in[4];
  const float* bk = (const float*)d_in[5];
  const float* bv = (const float*)d_in[6];
  float* out = (float*)d_out;

  const int M = BB * SS;                 // 8192 projection rows
  char* ws = (char*)d_ws;
  size_t off = 0;
  auto take = [&](size_t bytes) { char* p = ws + off; off += (bytes + 255) & ~(size_t)255; return p; };

  __bf16* xb  = (__bf16*)take((size_t)M * DD * 2);        // 8 MB
  __bf16* wqT = (__bf16*)take((size_t)DD * DD * 2);       // 0.5 MB each
  __bf16* wkT = (__bf16*)take((size_t)DD * DD * 2);
  __bf16* wvT = (__bf16*)take((size_t)DD * DD * 2);
  __bf16* Qb  = (__bf16*)take((size_t)M * DD * 2);        // 8 MB  [b][s][d]
  __bf16* Kb  = (__bf16*)take((size_t)M * DD * 2);        // 8 MB  [b][s][d]
  __bf16* VbT = (__bf16*)take((size_t)M * DD * 2);        // 8 MB  [b][d][s]
  float*  Sc  = (float*) take((size_t)BB * SS * SS * 4);  // 64 MB
  __bf16* Pb  = (__bf16*)take((size_t)BB * SS * SS * 2);  // 32 MB
  (void)ws_size;

  // k0: converts (+ weight transpose)
  {
    int nx = M * DD;
    k_cvt_bf16<<<(nx + 255) / 256, 256, 0, stream>>>(x, xb, nx);
    int nw = DD * DD;
    k_cvt_bf16_T<<<(nw + 255) / 256, 256, 0, stream>>>(Wq, wqT);
    k_cvt_bf16_T<<<(nw + 255) / 256, 256, 0, stream>>>(Wk, wkT);
    k_cvt_bf16_T<<<(nw + 255) / 256, 256, 0, stream>>>(Wv, wvT);
  }

  // k1: projections. waves = (M/32)*(DD/32) = 4096 -> 512 blocks
  {
    dim3 grid((M / 32) * (DD / 32) / 8);
    k_proj<false><<<grid, 256, 0, stream>>>(xb, wqT, bq, Qb);
    k_proj<false><<<grid, 256, 0, stream>>>(xb, wkT, bk, Kb);
    k_proj<true ><<<grid, 256, 0, stream>>>(xb, wvT, bv, VbT);
  }

  // k2: scores. per batch waves = (S/64)*(S/32) = 32*64 = 2048 -> 256 blocks
  {
    dim3 grid((SS / 64) * (SS / 32) / 8, BB);
    k_scores<<<grid, 256, 0, stream>>>(Qb, Kb, Sc);
  }

  // k3: softmax over B*S rows
  k_softmax<<<BB * SS, 256, 0, stream>>>(Sc, Pb);

  // k4: out = attn @ V. per batch waves = (S/64)*(D/32) = 512 -> 64 blocks
  {
    dim3 grid((SS / 64) * (DD / 32) / 8, BB);
    k_out<<<grid, 256, 0, stream>>>(Pb, VbT, out);
  }
  (void)in_sizes; (void)n_in; (void)out_size;
}
